// VisionTritonAttention_51118700757019
// MI455X (gfx1250) — matrix operand order repaired and checked by reading, not compiled, not run on hardware
//
#include <hip/hip_runtime.h>

typedef _Float16 v16h __attribute__((ext_vector_type(16)));
typedef _Float16 v8h  __attribute__((ext_vector_type(8)));
typedef float    v8f  __attribute__((ext_vector_type(8)));

#define HH 16     // heads
#define DD 128    // head dim
#define BM 64     // query rows per block (16 per wave)
#define BN 32     // keys per iteration
#define NW 4      // waves per block
#define PP 40     // P staging pitch (halves), padded vs bank conflicts
#define VP 40     // V staging pitch (halves), fallback kernel only

__device__ __forceinline__ v16h ldg_v16h(const _Float16* p) {
    v8h a = *(const v8h*)p;
    v8h b = *(const v8h*)(p + 8);
    return __builtin_shufflevector(a, b, 0,1,2,3,4,5,6,7,8,9,10,11,12,13,14,15);
}

// ============================================================================
// Pre-pass 1: K fp32 -> f16, natural [tok][h][d] layout
// ============================================================================
__global__ __launch_bounds__(256)
void convert_k_f16(const float* __restrict__ k, _Float16* __restrict__ kh, int n)
{
    const int i = (blockIdx.x * 256 + threadIdx.x) * 8;
    if (i + 7 >= n) {
        for (int t = 0; t < 8 && i + t < n; ++t) kh[i + t] = (_Float16)k[i + t];
        return;
    }
    float4 a = *(const float4*)(k + i);
    float4 b = *(const float4*)(k + i + 4);
    v8h h;
    h[0] = (_Float16)a.x; h[1] = (_Float16)a.y; h[2] = (_Float16)a.z; h[3] = (_Float16)a.w;
    h[4] = (_Float16)b.x; h[5] = (_Float16)b.y; h[6] = (_Float16)b.z; h[7] = (_Float16)b.w;
    *(v8h*)(kh + i) = h;
}

// ============================================================================
// Pre-pass 2: V fp32 [tok][h][d] -> f16 transposed [h][d][tok]
// LDS-tiled so both global read and write are coalesced.
// ============================================================================
__global__ __launch_bounds__(256)
void convert_v_tr_f16(const float* __restrict__ v, _Float16* __restrict__ vt, int T)
{
    __shared__ _Float16 tile[DD * 66];        // 128 x 64(+2 pad) halves
    const int hh   = blockIdx.y;
    const int tok0 = blockIdx.x * 64;
    const int tid  = threadIdx.x;

    #pragma unroll
    for (int i = 0; i < 32; ++i) {            // read: consecutive tid -> consecutive d
        const int idx = tid + 256 * i;        // 64 toks * 128 dims
        const int tok = idx >> 7, d = idx & 127;
        if (tok0 + tok < T)
            tile[d * 66 + tok] = (_Float16)v[((size_t)(tok0 + tok) * HH + hh) * DD + d];
    }
    __syncthreads();
    #pragma unroll
    for (int i = 0; i < 32; ++i) {            // write: consecutive tid -> consecutive tok
        const int idx = tid + 256 * i;
        const int d = idx >> 6, tok = idx & 63;
        if (tok0 + tok < T)
            vt[(size_t)(hh * DD + d) * T + tok0 + tok] = tile[d * 66 + tok];
    }
}

// ============================================================================
// Fast path: K f16 natural, V f16 pre-transposed -> no cvt / no V-LDS / no
// barriers in the hot loop. Pure fragment loads + WMMA + softmax.
// ============================================================================
__global__ __launch_bounds__(128)
void fa_varlen_wmma_f16(const float*    __restrict__ q,
                        const _Float16* __restrict__ kh,
                        const _Float16* __restrict__ vt,
                        const int*      __restrict__ cu_seqlens,
                        float*          __restrict__ out, int T)
{
    __shared__ _Float16 plds[NW * 16 * PP];   // per-wave P staging only (5120 B)

    const int b  = blockIdx.y;
    const int hh = blockIdx.z;
    const int cu  = cu_seqlens[b];
    const int len = cu_seqlens[b + 1] - cu;
    const int q0  = blockIdx.x * BM;
    if (q0 >= len) return;

    const int tid  = threadIdx.x;
    const int wave = tid >> 5;
    const int lane = tid & 31;
    const int g    = lane >> 4;
    const int ln   = lane & 15;

    const float scale = 0.08838834764831845f; // 1/sqrt(128)
    const float L2E   = 1.44269504088896340f;
    const int qrow0 = q0 + wave * 16;

    // Q tile: fp32 -> f16 A-fragments once per block (amortized conversion)
    v16h qa[4];
    {
        const float* qp = q + ((size_t)(cu + qrow0 + ln) * HH + hh) * DD;
        #pragma unroll
        for (int d = 0; d < 4; ++d) {
            #pragma unroll
            for (int j = 0; j < 2; ++j) {
                float tmp[8];
                *(float4*)&tmp[0] = *(const float4*)(qp + 32*d + 16*g + 8*j);
                *(float4*)&tmp[4] = *(const float4*)(qp + 32*d + 16*g + 8*j + 4);
                #pragma unroll
                for (int t = 0; t < 8; ++t)
                    qa[d][8*j + t] = (_Float16)(tmp[t] * scale);
            }
        }
    }

    float mrow[8], lrow[8];
    v8f o[8] = {};
    #pragma unroll
    for (int r = 0; r < 8; ++r) { mrow[r] = -3.0e30f; lrow[r] = 0.0f; }

    _Float16* pw = plds + wave * 16 * PP;
    const _Float16* vthead = vt + (size_t)hh * DD * T;
    const int ktiles = len / BN;

    for (int kt = 0; kt < ktiles; ++kt) {
        const int krow = cu + kt * BN;

        if (kt + 1 < ktiles) {                // global_prefetch_b8 next tiles
            __builtin_prefetch(kh + ((size_t)(krow + BN + lane) * HH + hh) * DD, 0, 1);
            __builtin_prefetch(vthead + (size_t)(lane * 4) * T + krow + BN, 0, 1);
        }

        // S = Q K^T : straight f16 fragment loads (16 contiguous halves each)
        v8f sc[2] = {};
        #pragma unroll
        for (int j2 = 0; j2 < 2; ++j2) {
            const _Float16* kp = kh + ((size_t)(krow + 16*j2 + ln) * HH + hh) * DD + 16*g;
            #pragma unroll
            for (int d = 0; d < 4; ++d) {
                v16h kb = ldg_v16h(kp + 32 * d);
                sc[j2] = __builtin_amdgcn_wmma_f32_16x16x32_f16(
                    false, qa[d], false, kb, (short)0, sc[j2], false, false);
            }
        }

        // online softmax
        float mnew[8], alpha[8];
        #pragma unroll
        for (int r = 0; r < 8; ++r) {
            float t = fmaxf(sc[0][r], sc[1][r]);
            t = fmaxf(t, __shfl_xor(t, 1, 32));
            t = fmaxf(t, __shfl_xor(t, 2, 32));
            t = fmaxf(t, __shfl_xor(t, 4, 32));
            t = fmaxf(t, __shfl_xor(t, 8, 32));
            mnew[r]  = fmaxf(mrow[r], t);
            alpha[r] = __builtin_amdgcn_exp2f((mrow[r] - mnew[r]) * L2E);
            mrow[r]  = mnew[r];
        }
        #pragma unroll
        for (int r = 0; r < 8; ++r) {
            float p0 = __builtin_amdgcn_exp2f((sc[0][r] - mnew[r]) * L2E);
            float p1 = __builtin_amdgcn_exp2f((sc[1][r] - mnew[r]) * L2E);
            pw[(r + 8*g) * PP + ln]      = (_Float16)p0;
            pw[(r + 8*g) * PP + 16 + ln] = (_Float16)p1;
            float s = p0 + p1;
            s += __shfl_xor(s, 1, 32);
            s += __shfl_xor(s, 2, 32);
            s += __shfl_xor(s, 4, 32);
            s += __shfl_xor(s, 8, 32);
            lrow[r] = lrow[r] * alpha[r] + s;
        }

        // P back in A layout (per-wave LDS, contiguous k = 16*g + s)
        v8h plo = *(const v8h*)(pw + ln * PP + 16*g);
        v8h phi = *(const v8h*)(pw + ln * PP + 16*g + 8);
        v16h pa = __builtin_shufflevector(plo, phi,
            0,1,2,3,4,5,6,7,8,9,10,11,12,13,14,15);

        #pragma unroll
        for (int dt = 0; dt < 8; ++dt) {
            #pragma unroll
            for (int r = 0; r < 8; ++r) o[dt][r] *= alpha[r];
        }
        // O += P * V : V fragments straight from pre-transposed global f16
        #pragma unroll
        for (int dt = 0; dt < 8; ++dt) {
            v16h vb = ldg_v16h(vthead + (size_t)(16*dt + ln) * T + krow + 16*g);
            o[dt] = __builtin_amdgcn_wmma_f32_16x16x32_f16(
                false, pa, false, vb, (short)0, o[dt], false, false);
        }
    }

    #pragma unroll
    for (int r = 0; r < 8; ++r) {
        const float inv = 1.0f / lrow[r];
        float* op = out + ((size_t)(cu + qrow0 + r + 8*g) * HH + hh) * DD + ln;
        #pragma unroll
        for (int dt = 0; dt < 8; ++dt)
            op[16 * dt] = o[dt][r] * inv;
    }
}

// ============================================================================
// Fallback (no workspace): fp32-direct kernel from previous round
// ============================================================================
__global__ __launch_bounds__(128)
void fa_varlen_wmma_f32(const float* __restrict__ q,
                        const float* __restrict__ k,
                        const float* __restrict__ v,
                        const int*   __restrict__ cu_seqlens,
                        float*       __restrict__ out)
{
    __shared__ _Float16 vlds[DD * VP];
    __shared__ _Float16 plds[NW * 16 * PP];

    const int b  = blockIdx.y;
    const int hh = blockIdx.z;
    const int cu  = cu_seqlens[b];
    const int len = cu_seqlens[b + 1] - cu;
    const int q0  = blockIdx.x * BM;
    if (q0 >= len) return;

    const int tid  = threadIdx.x;
    const int wave = tid >> 5;
    const int lane = tid & 31;
    const int g    = lane >> 4;
    const int ln   = lane & 15;

    const float scale = 0.08838834764831845f;
    const float L2E   = 1.44269504088896340f;
    const int qrow0 = q0 + wave * 16;

    v16h qa[4];
    {
        const float* qp = q + ((size_t)(cu + qrow0 + ln) * HH + hh) * DD;
        #pragma unroll
        for (int d = 0; d < 4; ++d) {
            #pragma unroll
            for (int j = 0; j < 2; ++j) {
                float tmp[8];
                *(float4*)&tmp[0] = *(const float4*)(qp + 32*d + 16*g + 8*j);
                *(float4*)&tmp[4] = *(const float4*)(qp + 32*d + 16*g + 8*j + 4);
                #pragma unroll
                for (int t = 0; t < 8; ++t)
                    qa[d][8*j + t] = (_Float16)(tmp[t] * scale);
            }
        }
    }

    float mrow[8], lrow[8];
    v8f o[8] = {};
    #pragma unroll
    for (int r = 0; r < 8; ++r) { mrow[r] = -3.0e30f; lrow[r] = 0.0f; }

    _Float16* pw = plds + wave * 16 * PP;
    const int ktiles = len / BN;

    for (int kt = 0; kt < ktiles; ++kt) {
        const int krow = cu + kt * BN;

        __syncthreads();
        {
            const int key   = tid >> 2;
            const int chunk = tid & 3;
            const float* vp = v + ((size_t)(krow + key) * HH + hh) * DD + 32*chunk;
            #pragma unroll
            for (int i = 0; i < 32; i += 4) {
                float4 f = *(const float4*)(vp + i);
                const int db = 32*chunk + i;
                vlds[(db + 0) * VP + key] = (_Float16)f.x;
                vlds[(db + 1) * VP + key] = (_Float16)f.y;
                vlds[(db + 2) * VP + key] = (_Float16)f.z;
                vlds[(db + 3) * VP + key] = (_Float16)f.w;
            }
        }
        __syncthreads();

        v8f sc[2] = {};
        #pragma unroll
        for (int j2 = 0; j2 < 2; ++j2) {
            const float* kp = k + ((size_t)(krow + 16*j2 + ln) * HH + hh) * DD + 16*g;
            #pragma unroll
            for (int d = 0; d < 4; ++d) {
                float tmp[16];
                *(float4*)&tmp[0]  = *(const float4*)(kp + 32*d + 0);
                *(float4*)&tmp[4]  = *(const float4*)(kp + 32*d + 4);
                *(float4*)&tmp[8]  = *(const float4*)(kp + 32*d + 8);
                *(float4*)&tmp[12] = *(const float4*)(kp + 32*d + 12);
                v16h kb;
                #pragma unroll
                for (int e = 0; e < 16; ++e) kb[e] = (_Float16)tmp[e];
                sc[j2] = __builtin_amdgcn_wmma_f32_16x16x32_f16(
                    false, qa[d], false, kb, (short)0, sc[j2], false, false);
            }
        }

        float mnew[8], alpha[8];
        #pragma unroll
        for (int r = 0; r < 8; ++r) {
            float t = fmaxf(sc[0][r], sc[1][r]);
            t = fmaxf(t, __shfl_xor(t, 1, 32));
            t = fmaxf(t, __shfl_xor(t, 2, 32));
            t = fmaxf(t, __shfl_xor(t, 4, 32));
            t = fmaxf(t, __shfl_xor(t, 8, 32));
            mnew[r]  = fmaxf(mrow[r], t);
            alpha[r] = __builtin_amdgcn_exp2f((mrow[r] - mnew[r]) * L2E);
            mrow[r]  = mnew[r];
        }
        #pragma unroll
        for (int r = 0; r < 8; ++r) {
            float p0 = __builtin_amdgcn_exp2f((sc[0][r] - mnew[r]) * L2E);
            float p1 = __builtin_amdgcn_exp2f((sc[1][r] - mnew[r]) * L2E);
            pw[(r + 8*g) * PP + ln]      = (_Float16)p0;
            pw[(r + 8*g) * PP + 16 + ln] = (_Float16)p1;
            float s = p0 + p1;
            s += __shfl_xor(s, 1, 32);
            s += __shfl_xor(s, 2, 32);
            s += __shfl_xor(s, 4, 32);
            s += __shfl_xor(s, 8, 32);
            lrow[r] = lrow[r] * alpha[r] + s;
        }

        v8h plo = *(const v8h*)(pw + ln * PP + 16*g);
        v8h phi = *(const v8h*)(pw + ln * PP + 16*g + 8);
        v16h pa = __builtin_shufflevector(plo, phi,
            0,1,2,3,4,5,6,7,8,9,10,11,12,13,14,15);

        #pragma unroll
        for (int dt = 0; dt < 8; ++dt) {
            #pragma unroll
            for (int r = 0; r < 8; ++r) o[dt][r] *= alpha[r];
        }
        #pragma unroll
        for (int dt = 0; dt < 8; ++dt) {
            const _Float16* vp2 = vlds + (16*dt + ln) * VP + 16*g;
            v16h vb = ldg_v16h(vp2);
            o[dt] = __builtin_amdgcn_wmma_f32_16x16x32_f16(
                false, pa, false, vb, (short)0, o[dt], false, false);
        }
    }

    #pragma unroll
    for (int r = 0; r < 8; ++r) {
        const float inv = 1.0f / lrow[r];
        float* op = out + ((size_t)(cu + qrow0 + r + 8*g) * HH + hh) * DD + ln;
        #pragma unroll
        for (int dt = 0; dt < 8; ++dt)
            op[16 * dt] = o[dt][r] * inv;
    }
}

extern "C" void kernel_launch(void* const* d_in, const int* in_sizes, int n_in,
                              void* d_out, int out_size, void* d_ws, size_t ws_size,
                              hipStream_t stream) {
    const float* q  = (const float*)d_in[0];
    const float* k  = (const float*)d_in[1];
    const float* v  = (const float*)d_in[2];
    const int*   cu = (const int*)d_in[3];
    float* out = (float*)d_out;

    const int B = in_sizes[3] - 1;
    const int n = in_sizes[0];            // T * H * D
    const int T = n / (HH * DD);
    const int SMAX = 1024;
    dim3 grid(SMAX / BM, B, HH);

    const size_t need = 4 * (size_t)n;    // kh (2n bytes) + vt (2n bytes)
    if (ws_size >= need) {
        _Float16* kh = (_Float16*)d_ws;
        _Float16* vt = kh + n;
        convert_k_f16<<<(n / 8 + 255) / 256, 256, 0, stream>>>(k, kh, n);
        convert_v_tr_f16<<<dim3((T + 63) / 64, HH), 256, 0, stream>>>(v, vt, T);
        fa_varlen_wmma_f16<<<grid, NW * 32, 0, stream>>>(q, kh, vt, cu, out, T);
    } else {
        fa_varlen_wmma_f32<<<grid, NW * 32, 0, stream>>>(q, k, v, cu, out);
    }
}
